// GraphAttentionLayer_20633022890032
// MI455X (gfx1250) — matrix operand order repaired and checked by reading, not compiled, not run on hardware
//
#include <hip/hip_runtime.h>
#include <math.h>

typedef __attribute__((ext_vector_type(16))) _Float16 v16h;
typedef __attribute__((ext_vector_type(8)))  float    v8f;

#define D_IN  256
#define D_OUT 64

// ---------------------------------------------------------------------------
// Kernel 1: H = leaky_relu_0.01(X @ W) for both W_high and W_low.
// One 16x16 C tile per wave via v_wmma_f32_16x16x32_f16, f32 precision
// recovered with a 2-term f16 split (hi*hi + hi*lo + lo*hi).
// block = 128 threads (4 waves -> 4 column tiles = full D_OUT=64)
// grid  = (ceil(N/16), 2)  [y: 0=high, 1=low]
// ---------------------------------------------------------------------------
__global__ __launch_bounds__(128) void gat_gemm_kernel(
    const float* __restrict__ X,
    const float* __restrict__ Wh,
    const float* __restrict__ Wl,
    float* __restrict__ h_high,
    float* __restrict__ h_low,
    int Nn)
{
    const int lane = threadIdx.x & 31;
    const int wave = threadIdx.x >> 5;          // column tile 0..3
    const int m0   = blockIdx.x * 16;
    const int n0   = wave * 16;
    const float* __restrict__ Wp = blockIdx.y ? Wl : Wh;
    float* __restrict__ Hp       = blockIdx.y ? h_low : h_high;

    const int mrow = lane & 15;
    const int half = lane >> 4;

    int arow_i = m0 + mrow;
    if (arow_i >= Nn) arow_i = Nn - 1;          // clamp (N%16==0 normally)

    v8f acc = {};

    for (int k0 = 0; k0 < D_IN; k0 += 32) {
        // ---- A fragment: lane holds row (m0+mrow); element e -> K=k0+half*16+e
        const float* ap = X + (size_t)arow_i * D_IN + k0 + half * 16;
        float4 a0 = *(const float4*)(ap);
        float4 a1 = *(const float4*)(ap + 4);
        float4 a2 = *(const float4*)(ap + 8);
        float4 a3 = *(const float4*)(ap + 12);
        float af[16] = { a0.x, a0.y, a0.z, a0.w, a1.x, a1.y, a1.z, a1.w,
                         a2.x, a2.y, a2.z, a2.w, a3.x, a3.y, a3.z, a3.w };
        v16h ahi, alo;
        #pragma unroll
        for (int t = 0; t < 16; ++t) {
            _Float16 hv = (_Float16)af[t];
            ahi[t] = hv;
            alo[t] = (_Float16)(af[t] - (float)hv);
        }

        // ---- B fragment: lane holds col (n0+mrow); element e -> K=k0+half*16+e
        const float* bp = Wp + (size_t)(k0 + half * 16) * D_OUT + (n0 + mrow);
        v16h bhi, blo;
        #pragma unroll
        for (int t = 0; t < 16; ++t) {
            float b = bp[(size_t)t * D_OUT];
            _Float16 hv = (_Float16)b;
            bhi[t] = hv;
            blo[t] = (_Float16)(b - (float)hv);
        }

        acc = __builtin_amdgcn_wmma_f32_16x16x32_f16(false, ahi, false, bhi,
                                                     (short)0, acc, false, false);
        acc = __builtin_amdgcn_wmma_f32_16x16x32_f16(false, ahi, false, blo,
                                                     (short)0, acc, false, false);
        acc = __builtin_amdgcn_wmma_f32_16x16x32_f16(false, alo, false, bhi,
                                                     (short)0, acc, false, false);
    }

    // C/D layout: element v -> row m0 + v + 8*half, col n0 + (lane&15)
    const int col = n0 + mrow;
    float* __restrict__ outp = Hp + (size_t)(m0 + 8 * half) * D_OUT + col;
    if (m0 + 15 < Nn) {
        // full tile: straight-line, unpredicated stores (hot path; N%16==0)
        #pragma unroll
        for (int v = 0; v < 8; ++v) {
            float x = acc[v];
            outp[(size_t)v * D_OUT] = (x >= 0.0f) ? x : 0.01f * x;
        }
    } else {
        #pragma unroll
        for (int v = 0; v < 8; ++v) {
            int row = m0 + v + 8 * half;
            if (row < Nn) {
                float x = acc[v];
                Hp[(size_t)row * D_OUT + col] = (x >= 0.0f) ? x : 0.01f * x;
            }
        }
    }
}

// ---------------------------------------------------------------------------
// Kernel 2: per-node scalar projections s = h . a  (one wave per node)
// ---------------------------------------------------------------------------
__global__ __launch_bounds__(256) void gat_proj_kernel(
    const float* __restrict__ h_high, const float* __restrict__ h_low,
    const float* __restrict__ a_high, const float* __restrict__ a_low,
    float* __restrict__ s_high, float* __restrict__ s_low, int Nn)
{
    const int lane = threadIdx.x & 31;
    const int node = blockIdx.x * (blockDim.x >> 5) + (threadIdx.x >> 5);
    if (node >= Nn) return;

    const float2* hh = (const float2*)(h_high + (size_t)node * D_OUT);
    const float2* hl = (const float2*)(h_low  + (size_t)node * D_OUT);
    const float2* ah = (const float2*)a_high;
    const float2* al = (const float2*)a_low;

    float2 vh = hh[lane], vl = hl[lane];
    float2 ch = ah[lane], cl = al[lane];
    float ph = vh.x * ch.x + vh.y * ch.y;
    float pl = vl.x * cl.x + vl.y * cl.y;

    #pragma unroll
    for (int off = 16; off > 0; off >>= 1) {
        ph += __shfl_xor(ph, off, 32);
        pl += __shfl_xor(pl, off, 32);
    }
    if (lane == 0) { s_high[node] = ph; s_low[node] = pl; }
}

// ---------------------------------------------------------------------------
// Kernel 3: edge scatter. Each wave owns 32 edges: lanes compute the 32
// attention weights in parallel, then for each edge all 32 lanes scatter the
// 64-dim (float2/lane) weighted feature with global_atomic_add_f32.
// ---------------------------------------------------------------------------
__global__ __launch_bounds__(256) void gat_edge_kernel(
    const int* __restrict__ esrc, const int* __restrict__ edst,
    const float* __restrict__ s_high, const float* __restrict__ s_low,
    const float* __restrict__ h_high, const float* __restrict__ h_low,
    float* __restrict__ hp_high, float* __restrict__ hp_low,
    float* __restrict__ rs_high, float* __restrict__ rs_low, int E)
{
    const int lane = threadIdx.x & 31;
    const long wave = (long)blockIdx.x * (blockDim.x >> 5) + (threadIdx.x >> 5);
    const long base = wave * 32;
    if (base >= E) return;

    int e = (int)base + lane;
    float wh = 0.0f, wl = 0.0f;
    int si = 0, di = 0;
    if (e < E) {
        si = esrc[e];
        di = edst[e];
        float sh = s_high[si] - s_high[di];
        sh = (sh >= 0.0f) ? sh : 0.2f * sh;          // leaky(0.2) on logit
        float sl = s_low[si] + s_low[di];
        sl = (sl >= 0.0f) ? sl : 0.2f * sl;
        wh = __expf(-sh);
        wl = __expf(-sl);
        atomicAdd(&rs_high[si], wh);
        atomicAdd(&rs_low[si],  wl);
    }

    int cnt = (int)(((long)E - base) < 32 ? ((long)E - base) : 32);
    const float2* H2 = (const float2*)h_high;
    const float2* L2p = (const float2*)h_low;

    for (int i = 0; i < cnt; ++i) {
        float w_h = __shfl(wh, i, 32);
        float w_l = __shfl(wl, i, 32);
        int   d   = __shfl(di, i, 32);
        int   s   = __shfl(si, i, 32);

        float2 hv = H2[(size_t)d * 32 + lane];
        float2 lv = L2p[(size_t)d * 32 + lane];

        float* ph = hp_high + (size_t)s * D_OUT + lane * 2;
        atomicAdd(ph,     w_h * hv.x);
        atomicAdd(ph + 1, w_h * hv.y);
        float* pl = hp_low + (size_t)s * D_OUT + lane * 2;
        atomicAdd(pl,     w_l * lv.x);
        atomicAdd(pl + 1, w_l * lv.y);
    }
}

// ---------------------------------------------------------------------------
// Kernel 4: out = leaky_0.01( hp / (rowsum + 1e-16) ), concat layout [N,128]
// ---------------------------------------------------------------------------
__global__ __launch_bounds__(256) void gat_final_kernel(
    const float* __restrict__ hp_high, const float* __restrict__ hp_low,
    const float* __restrict__ rs_high, const float* __restrict__ rs_low,
    float* __restrict__ out, int total)
{
    int i = blockIdx.x * blockDim.x + threadIdx.x;
    if (i >= total) return;
    int n = i >> 7;
    int c = i & 127;
    float v;
    if (c < D_OUT)
        v = hp_high[(size_t)n * D_OUT + c] / (rs_high[n] + 1e-16f);
    else
        v = hp_low[(size_t)n * D_OUT + (c - D_OUT)] / (rs_low[n] + 1e-16f);
    out[i] = (v >= 0.0f) ? v : 0.01f * v;
}

// ---------------------------------------------------------------------------
extern "C" void kernel_launch(void* const* d_in, const int* in_sizes, int n_in,
                              void* d_out, int out_size, void* d_ws, size_t ws_size,
                              hipStream_t stream)
{
    const float* X    = (const float*)d_in[0];
    const int*   edge = (const int*)  d_in[1];
    const float* Wh   = (const float*)d_in[2];
    const float* Wl   = (const float*)d_in[3];
    const float* ah   = (const float*)d_in[4];
    const float* al   = (const float*)d_in[5];

    const int Nn = in_sizes[0] / D_IN;       // 50000
    const int E  = in_sizes[1] / 2;          // 1600000
    const int* esrc = edge;
    const int* edst = edge + E;

    float* ws = (float*)d_ws;
    size_t o = 0;
    float* h_high = ws + o; o += (size_t)Nn * D_OUT;
    float* h_low  = ws + o; o += (size_t)Nn * D_OUT;
    float* s_high = ws + o; o += (size_t)Nn;
    float* s_low  = ws + o; o += (size_t)Nn;
    float* accum  = ws + o;                  // hp_high | hp_low | rs_high | rs_low
    float* hp_high = accum;
    float* hp_low  = accum + (size_t)Nn * D_OUT;
    float* rs_high = accum + (size_t)Nn * D_OUT * 2;
    float* rs_low  = rs_high + Nn;

    // zero the atomic accumulators (hp + rowsums) each call
    hipMemsetAsync(accum, 0, sizeof(float) * ((size_t)Nn * (2 * D_OUT + 2)), stream);

    // 1) GEMM + leaky (WMMA f16-split, f32 accurate)
    dim3 g1((Nn + 15) / 16, 2);
    gat_gemm_kernel<<<g1, 128, 0, stream>>>(X, Wh, Wl, h_high, h_low, Nn);

    // 2) per-node projections
    int projBlocks = (Nn + 7) / 8;
    gat_proj_kernel<<<projBlocks, 256, 0, stream>>>(h_high, h_low, ah, al,
                                                    s_high, s_low, Nn);

    // 3) edge scatter (32 edges per wave, 8 waves per block)
    long waveTasks = ((long)E + 31) / 32;
    int edgeBlocks = (int)((waveTasks + 7) / 8);
    gat_edge_kernel<<<edgeBlocks, 256, 0, stream>>>(esrc, edst, s_high, s_low,
                                                    h_high, h_low, hp_high, hp_low,
                                                    rs_high, rs_low, E);

    // 4) normalize + final leaky -> d_out [N,128]
    int total = Nn * 2 * D_OUT;
    gat_final_kernel<<<(total + 255) / 256, 256, 0, stream>>>(hp_high, hp_low,
                                                              rs_high, rs_low,
                                                              (float*)d_out, total);
}
